// QuantLinear_26534307954843
// MI455X (gfx1250) — matrix operand order repaired and checked by reading, not compiled, not run on hardware
//
#include <hip/hip_runtime.h>

typedef __attribute__((ext_vector_type(2)))  _Float16 v2h;
typedef __attribute__((ext_vector_type(4)))  _Float16 v4h;
typedef __attribute__((ext_vector_type(8)))  _Float16 v8h;
typedef __attribute__((ext_vector_type(16))) _Float16 v16h;
typedef __attribute__((ext_vector_type(8)))  float    v8f;
typedef __attribute__((ext_vector_type(4)))  float    v4f;
typedef __attribute__((ext_vector_type(4)))  int      v4i;

#define M_TOTAL 8192   // B*S
#define N_TOTAL 4096   // OUT
#define K_TOTAL 4096   // IN
#define GS      128

#define MT  256        // workgroup M tile (4 waves x 64 rows)
#define NT  128        // workgroup N tile (2 waves x 64 cols)
#define KT  32         // K chunk = one WMMA k-step
#define LDK 40         // padded halfs per LDS row (80B stride -> conflict-free frags)

// v_cvt_pkrtz_f16_f32 returns a __fp16 vector; bit-cast to our _Float16 vector type.
static __device__ __forceinline__ v2h cvt_pk_h2(float a, float b) {
    return __builtin_bit_cast(v2h, __builtin_amdgcn_cvt_pkrtz(a, b));
}

__global__ __launch_bounds__(256)
void nf4_wmma_gemm(const float* __restrict__ x,
                   const int*   __restrict__ qweight,   // [N, K/2], one packed byte per int32
                   const float* __restrict__ scales,    // [N, K/GS]
                   const float* __restrict__ bias,      // [N]
                   const float* __restrict__ lut,       // [16]
                   float*       __restrict__ out)       // [M, N]
{
    __shared__ _Float16 lA[2][MT][LDK];   // x tile, f16          (2*256*40*2 = 40 KB)
    __shared__ _Float16 lB[2][NT][LDK];   // dequant weights, f16 (2*128*40*2 = 20 KB)
    __shared__ v2h      lutH[256];        // byte -> (lut[lo], lut[hi]) packed f16

    const int tid      = threadIdx.x;
    const int lane     = tid & 31;
    const int wid      = tid >> 5;
    const int wm       = wid & 3;        // wave M position: 4 waves * 64 rows = 256
    const int wn       = wid >> 2;       // wave N position: 2 waves * 64 cols = 128
    const int halfLane = lane >> 4;      // 0 for lanes 0-15, 1 for lanes 16-31
    const int l16      = lane & 15;

    const int tileM = blockIdx.y * MT;
    const int tileN = blockIdx.x * NT;

    // Build byte -> packed-half2 (lut[low nibble], lut[high nibble]) table.
    {
        const int lo = tid & 15;
        const int hi = (tid >> 4) & 15;
        lutH[tid] = cvt_pk_h2(lut[lo], lut[hi]);
    }

    // Accumulators: wave covers 64(M) x 64(N) = 4 x 4 tiles of 16x16 f32.
    v8f acc[4][4];
#pragma unroll
    for (int sm = 0; sm < 4; ++sm)
#pragma unroll
        for (int sn = 0; sn < 4; ++sn)
            acc[sm][sn] = (v8f){0.f, 0.f, 0.f, 0.f, 0.f, 0.f, 0.f, 0.f};

    // Prefetch registers (software pipeline).
    v4f  xr[8];   // 256x32 f32 x-tile: 256 thr * 8 * float4
    v4i  wr[2];   // 128x16 int32 qweight tile: 256 thr * 2 * int4
    float sc[2];  // per-row scale for the current K chunk

    auto loadX = [&](int kc) {
#pragma unroll
        for (int i = 0; i < 8; ++i) {
            const int s   = tid + i * 256;        // 0..2047 float4 slots
            const int row = s >> 3;               // 0..255
            const int c4  = (s & 7) * 4;          // 0..28
            xr[i] = *(const v4f*)(x + (size_t)(tileM + row) * K_TOTAL + kc + c4);
        }
    };

    auto loadW = [&](int kc) {
        const int g = kc / GS;                    // scale group, constant within chunk
#pragma unroll
        for (int i = 0; i < 2; ++i) {
            const int s   = tid + i * 256;        // 0..511 int4 slots
            const int row = s >> 2;               // 0..127
            const int c4  = (s & 3) * 4;          // 0..12 (int32 index within chunk)
            wr[i] = *(const v4i*)(qweight + (size_t)(tileN + row) * (K_TOTAL / 2)
                                  + (kc >> 1) + c4);
            sc[i] = scales[(size_t)(tileN + row) * (K_TOTAL / GS) + g];
        }
    };

    auto storeX = [&](int buf) {
#pragma unroll
        for (int i = 0; i < 8; ++i) {
            const int s   = tid + i * 256;
            const int row = s >> 3;
            const int c4  = (s & 7) * 4;
            const v2h p0  = cvt_pk_h2(xr[i].x, xr[i].y);
            const v2h p1  = cvt_pk_h2(xr[i].z, xr[i].w);
            *(v4h*)&lA[buf][row][c4] = __builtin_shufflevector(p0, p1, 0, 1, 2, 3);
        }
    };

    auto storeW = [&](int buf) {
#pragma unroll
        for (int i = 0; i < 2; ++i) {
            const int s   = tid + i * 256;
            const int row = s >> 2;
            const int c4  = (s & 3) * 4;          // base int index -> k pairs
            const _Float16 sh = (_Float16)sc[i];
            const v2h sv = {sh, sh};
            v2h w[4];
#pragma unroll
            for (int j = 0; j < 4; ++j)
                w[j] = lutH[wr[i][j] & 255] * sv;  // v_pk_mul_f16
            const v4h lo = __builtin_shufflevector(w[0], w[1], 0, 1, 2, 3);
            const v4h hi = __builtin_shufflevector(w[2], w[3], 0, 1, 2, 3);
            *(v8h*)&lB[buf][row][c4 * 2] =
                __builtin_shufflevector(lo, hi, 0, 1, 2, 3, 4, 5, 6, 7);
        }
    };

    auto compute = [&](int buf) {
        v16h a[4], b[4];
        // A fragments (16x32 f16): lane<16 -> K {0..7,16..23}; lane>=16 -> K {8..15,24..31}
#pragma unroll
        for (int sm = 0; sm < 4; ++sm) {
            const int row = wm * 64 + sm * 16 + l16;
            const v8h a0 = *(const v8h*)&lA[buf][row][halfLane * 8];
            const v8h a1 = *(const v8h*)&lA[buf][row][16 + halfLane * 8];
            a[sm] = __builtin_shufflevector(a0, a1, 0, 1, 2, 3, 4, 5, 6, 7,
                                            8, 9, 10, 11, 12, 13, 14, 15);
        }
        // B fragments (32x16 f16): lane<16 -> K {0..7,16..23}; lane>=16 -> K {8..15,24..31}
#pragma unroll
        for (int sn = 0; sn < 4; ++sn) {
            const int col = wn * 64 + sn * 16 + l16;
            const v8h b0 = *(const v8h*)&lB[buf][col][halfLane * 8];
            const v8h b1 = *(const v8h*)&lB[buf][col][16 + halfLane * 8];
            b[sn] = __builtin_shufflevector(b0, b1, 0, 1, 2, 3, 4, 5, 6, 7,
                                            8, 9, 10, 11, 12, 13, 14, 15);
        }
#pragma unroll
        for (int sm = 0; sm < 4; ++sm)
#pragma unroll
            for (int sn = 0; sn < 4; ++sn)
                acc[sm][sn] = __builtin_amdgcn_wmma_f32_16x16x32_f16(
                    false, a[sm], false, b[sn], (short)0, acc[sm][sn], false, false);
    };

    // Prologue: lut table must be visible before first dequant store.
    loadX(0);
    loadW(0);
    __syncthreads();        // lutH ready
    storeX(0);
    storeW(0);
    __syncthreads();        // buffer 0 ready

    const int NCH = K_TOTAL / KT;  // 128 chunks
    for (int ch = 0; ch < NCH; ++ch) {
        const int buf = ch & 1;
        if (ch + 1 < NCH) {
            loadX((ch + 1) * KT);
            loadW((ch + 1) * KT);
        }
        compute(buf);
        if (ch + 1 < NCH) {
            storeX(buf ^ 1);
            storeW(buf ^ 1);
        }
        __syncthreads();
    }

    // Epilogue: C/D layout -> lane l (<16): M=r, N=l ; lane>=16: M=r+8, N=l-16.
#pragma unroll
    for (int sn = 0; sn < 4; ++sn) {
        const int n    = tileN + wn * 64 + sn * 16 + l16;
        const float bn = bias[n];
#pragma unroll
        for (int sm = 0; sm < 4; ++sm) {
            const int mbase = tileM + wm * 64 + sm * 16 + halfLane * 8;
#pragma unroll
            for (int r = 0; r < 8; ++r) {
                out[(size_t)(mbase + r) * N_TOTAL + n] = acc[sm][sn][r] + bn;
            }
        }
    }
}

extern "C" void kernel_launch(void* const* d_in, const int* in_sizes, int n_in,
                              void* d_out, int out_size, void* d_ws, size_t ws_size,
                              hipStream_t stream) {
    (void)in_sizes; (void)n_in; (void)d_ws; (void)ws_size; (void)out_size;
    const float* x       = (const float*)d_in[0];
    const int*   qweight = (const int*)d_in[1];
    const float* scales  = (const float*)d_in[2];
    const float* bias    = (const float*)d_in[3];
    // d_in[4] = g_idx (implied by k/GS, unused), d_in[5] = lut
    const float* lut     = (const float*)d_in[5];
    float* out           = (float*)d_out;

    dim3 grid(N_TOTAL / NT, M_TOTAL / MT);   // 32 x 32 workgroups
    nf4_wmma_gemm<<<grid, 256, 0, stream>>>(x, qweight, scales, bias, lut, out);
}
